// Fracformer_33663953666359
// MI455X (gfx1250) — matrix operand order repaired and checked by reading, compiled, not run on hardware
//
#include <hip/hip_runtime.h>
#include <cstdint>
#include <cstddef>

// ---------------- problem constants ----------------
#define BB    4
#define NN    2048
#define NNK   2048
#define CC    384
#define HH    6
#define HD    64
#define KNN   8
#define HIDD  1536
#define MROWS (BB * NN)      // 8192 query rows (== BB*NNK value rows)
#define BHH   (BB * HH)      // 24 head-batches

// ---------------- WMMA plumbing (CDNA5 / gfx1250, wave32) ----------------
typedef __attribute__((ext_vector_type(16))) __bf16 v16bf;
typedef __attribute__((ext_vector_type(8)))  float  v8f;

#define WMMA_BF16(a, b, c) \
  __builtin_amdgcn_wmma_f32_16x16x32_bf16(false, (a), false, (b), (short)0, (c), false, false)

// A-matrix 16x32 bf16 fragment (ISA 7.12.2): lane lo = row, halves at kh and 16+kh, kh = 8*(lane>=16)
__device__ __forceinline__ v16bf load_frag_a(const __bf16* p, int kh) {
  union { v16bf v; uint4 q[2]; } f;
  f.q[0] = *(const uint4*)(p + kh);
  f.q[1] = *(const uint4*)(p + 16 + kh);
  return f.v;
}
// B-matrix 32x16 bf16 fragment: lane lo = column, halves at K rows 8*hi and 16+8*hi (same order as A)
__device__ __forceinline__ v16bf load_frag_b(const __bf16* p, int hi) {
  union { v16bf v; uint4 q[2]; } f;
  f.q[0] = *(const uint4*)(p + hi * 8);
  f.q[1] = *(const uint4*)(p + 16 + hi * 8);
  return f.v;
}

// async copy of 16 bytes/lane from global to LDS (gfx1250; tracked on ASYNCcnt)
__device__ __forceinline__ void async_copy_b128(const void* gsrc, void* lds_dst) {
  uint32_t l = (uint32_t)(uintptr_t)lds_dst;               // LDS offset = addr[31:0]
  unsigned long long g = (unsigned long long)(uintptr_t)gsrc;
  asm volatile("global_load_async_to_lds_b128 %0, %1, off" :: "v"(l), "v"(g) : "memory");
}
__device__ __forceinline__ void wait_async0() {
  asm volatile("s_wait_asynccnt 0x0" ::: "memory");
}

// C/D 16x16 f32 layout: lane lo = col, row = 8*hi + r
__device__ __forceinline__ void epilogue_tile(v8f c, int row0, int col0, int lo, int hi,
                                              const float* __restrict__ bias,
                                              float* __restrict__ outF,
                                              __bf16* __restrict__ outB,
                                              int ldo, int accum) {
  const int col = col0 + lo;
  const float bv = bias ? bias[col] : 0.0f;
#pragma unroll
  for (int r = 0; r < 8; ++r) {
    const size_t off = (size_t)(row0 + hi * 8 + r) * ldo + col;
    float val = c[r] + bv;
    if (accum) val += outF[off];
    if (outF) outF[off] = val;
    if (outB) outB[off] = (__bf16)val;
  }
}

// ---------------- generic bf16 GEMM: out[M,N] = A[M,K] @ W[N,K]^T (+bias) ----------------
// 8 waves/block; wave tile 32x32 (2x2 WMMA tiles); block tile 128(M) x 64(N).
// W tile (64x32) staged in LDS via async-LDS loads, double buffered; A direct + prefetch.
// NOTE: grids must tile M,N exactly (all launched shapes do).
__global__ __launch_bounds__(256) void gemm_bf16_k(
    const __bf16* __restrict__ A, int lda,
    const __bf16* __restrict__ W, int ldw,
    const float* __restrict__ bias,
    float* __restrict__ outF, __bf16* __restrict__ outB, int ldo,
    int M, int N, int Kd, int accum)
{
  __shared__ __align__(16) __bf16 Btile[2][64][32];   // 8 KB double-buffered weight tile
  const int tid  = threadIdx.x;
  const int w    = tid >> 5;
  const int lane = tid & 31;
  const int lo = lane & 15, hi = lane >> 4, kh = hi * 8;
  const int rowBlk  = blockIdx.x * 128 + (w & 3) * 32;
  const int colBlk0 = blockIdx.y * 64;
  const int colBlk  = colBlk0 + (w >> 2) * 32;

  // this thread's 16-byte chunk of the 64x32 W tile copy
  const int crow = tid >> 2;          // 0..63 : B column == W row
  const int cchk = (tid & 3) * 8;     // 0,8,16,24 : element offset in k strip
  const __bf16* wsrc = W + (size_t)(colBlk0 + crow) * ldw + cchk;

  const __bf16* a0p = A + (size_t)(rowBlk + lo) * lda;
  const __bf16* a1p = A + (size_t)(rowBlk + 16 + lo) * lda;
  const __bf16* bl0 = &Btile[0][(w >> 2) * 32 + lo][0];   // this wave's B columns in LDS
  const __bf16* bl1 = &Btile[1][(w >> 2) * 32 + lo][0];

  async_copy_b128(wsrc, &Btile[0][crow][cchk]);           // prologue: first W tile

  v8f c00 = {}, c01 = {}, c10 = {}, c11 = {};
  const int steps = Kd >> 5;
  for (int s = 0; s < steps; ++s) {
    const int k0 = s << 5;
    wait_async0();                     // own async chunk for buf (s&1) arrived
    __syncthreads();                   // all waves' chunks visible
    if (s + 1 < steps)                 // overlap next tile copy with compute
      async_copy_b128(wsrc + k0 + 32, &Btile[(s + 1) & 1][crow][cchk]);
    if (s + 2 < steps) {               // prefetch A two steps ahead
      __builtin_prefetch(a0p + k0 + 64, 0, 0);
      __builtin_prefetch(a1p + k0 + 64, 0, 0);
    }
    const __bf16* bp = (s & 1) ? bl1 : bl0;
    v16bf a0 = load_frag_a(a0p + k0, kh);
    v16bf a1 = load_frag_a(a1p + k0, kh);
    v16bf b0 = load_frag_b(bp, hi);             // cols colBlk .. +15
    v16bf b1 = load_frag_b(bp + 16 * 32, hi);   // cols colBlk+16 .. +31
    c00 = WMMA_BF16(a0, b0, c00);
    c01 = WMMA_BF16(a0, b1, c01);
    c10 = WMMA_BF16(a1, b0, c10);
    c11 = WMMA_BF16(a1, b1, c11);
  }
  epilogue_tile(c00, rowBlk,      colBlk,      lo, hi, bias, outF, outB, ldo, accum);
  epilogue_tile(c01, rowBlk,      colBlk + 16, lo, hi, bias, outF, outB, ldo, accum);
  epilogue_tile(c10, rowBlk + 16, colBlk,      lo, hi, bias, outF, outB, ldo, accum);
  epilogue_tile(c11, rowBlk + 16, colBlk + 16, lo, hi, bias, outF, outB, ldo, accum);
}

// ---------------- flash attention (one wave = 16 query rows, hd = 64) ----------------
// Qh/Kh: [BH][Nq|Nk][64] bf16 ; Vt: [BH][64][Nk] bf16 ; out: [B*Nq, CC] bf16 at col h*64
__global__ __launch_bounds__(256) void flash_attn_k(
    const __bf16* __restrict__ Qh, const __bf16* __restrict__ Kh,
    const __bf16* __restrict__ Vt, __bf16* __restrict__ outB,
    int Nq, int Nk, float scale)
{
  __shared__ __align__(16) __bf16 Pld[8][16][32];  // per-wave P tile scratch
  const int w    = threadIdx.x >> 5;
  const int lane = threadIdx.x & 31;
  const int lo = lane & 15, hi = lane >> 4, kh = hi * 8;
  const int bh = blockIdx.y, b = bh / HH, h = bh % HH;
  const int q0 = blockIdx.x * 128 + w * 16;

  const __bf16* Qp = Qh + ((size_t)bh * Nq + q0 + lo) * HD;
  const v16bf aQ0 = load_frag_a(Qp, kh);        // d = 0..31
  const v16bf aQ1 = load_frag_a(Qp + 32, kh);   // d = 32..63
  const __bf16* Kbh = Kh + (size_t)bh * Nk * HD;
  const __bf16* Vbh = Vt + (size_t)bh * HD * Nk;

  v8f o0 = {}, o1 = {}, o2 = {}, o3 = {};
  float rm[8], rl[8];
#pragma unroll
  for (int r = 0; r < 8; ++r) { rm[r] = -1e30f; rl[r] = 0.0f; }

  for (int n0 = 0; n0 < Nk; n0 += 32) {
    const __bf16* kp0 = Kbh + (size_t)(n0 + lo) * HD;
    const __bf16* kp1 = Kbh + (size_t)(n0 + 16 + lo) * HD;
    v8f s0 = {}, s1 = {};
    s0 = WMMA_BF16(aQ0, load_frag_b(kp0, hi), s0);
    s1 = WMMA_BF16(aQ0, load_frag_b(kp1, hi), s1);
    s0 = WMMA_BF16(aQ1, load_frag_b(kp0 + 32, hi), s0);
    s1 = WMMA_BF16(aQ1, load_frag_b(kp1 + 32, hi), s1);

    float alpha[8];
#pragma unroll
    for (int r = 0; r < 8; ++r) {
      float x0 = s0[r] * scale, x1 = s1[r] * scale;
      float mx = fmaxf(x0, x1);                       // row reduce over 16-lane group
      mx = fmaxf(mx, __shfl_xor(mx, 1, 32));
      mx = fmaxf(mx, __shfl_xor(mx, 2, 32));
      mx = fmaxf(mx, __shfl_xor(mx, 4, 32));
      mx = fmaxf(mx, __shfl_xor(mx, 8, 32));
      float mn = fmaxf(rm[r], mx);
      float p0 = __expf(x0 - mn), p1 = __expf(x1 - mn);
      float sum = p0 + p1;
      sum += __shfl_xor(sum, 1, 32);
      sum += __shfl_xor(sum, 2, 32);
      sum += __shfl_xor(sum, 4, 32);
      sum += __shfl_xor(sum, 8, 32);
      float al = __expf(rm[r] - mn);
      rl[r] = rl[r] * al + sum;
      rm[r] = mn;
      alpha[r] = al;
      Pld[w][hi * 8 + r][lo]      = (__bf16)p0;       // C layout -> LDS
      Pld[w][hi * 8 + r][16 + lo] = (__bf16)p1;
    }
#pragma unroll
    for (int r = 0; r < 8; ++r) {
      o0[r] *= alpha[r]; o1[r] *= alpha[r]; o2[r] *= alpha[r]; o3[r] *= alpha[r];
    }
    v16bf aP = load_frag_a(&Pld[w][lo][0], kh);       // re-read as A fragment
    o0 = WMMA_BF16(aP, load_frag_b(Vbh + (size_t)(0 * 16 + lo) * Nk + n0, hi), o0);
    o1 = WMMA_BF16(aP, load_frag_b(Vbh + (size_t)(1 * 16 + lo) * Nk + n0, hi), o1);
    o2 = WMMA_BF16(aP, load_frag_b(Vbh + (size_t)(2 * 16 + lo) * Nk + n0, hi), o2);
    o3 = WMMA_BF16(aP, load_frag_b(Vbh + (size_t)(3 * 16 + lo) * Nk + n0, hi), o3);
  }

#pragma unroll
  for (int r = 0; r < 8; ++r) {
    float inv = 1.0f / rl[r];
    size_t off = ((size_t)b * Nq + q0 + hi * 8 + r) * CC + h * HD;
    outB[off + lo]      = (__bf16)(o0[r] * inv);
    outB[off + 16 + lo] = (__bf16)(o1[r] * inv);
    outB[off + 32 + lo] = (__bf16)(o2[r] * inv);
    outB[off + 48 + lo] = (__bf16)(o3[r] * inv);
  }
}

// ---------------- elementwise helpers ----------------
__global__ __launch_bounds__(128) void layernorm_k(
    const float* __restrict__ x, const float* __restrict__ w,
    const float* __restrict__ b, __bf16* __restrict__ out)
{
  const int row = blockIdx.x, tid = threadIdx.x;
  const float* xr = x + (size_t)row * CC;
  __shared__ float red[128];
  float s = 0.f;
  for (int c = tid; c < CC; c += 128) s += xr[c];
  red[tid] = s; __syncthreads();
  for (int o = 64; o > 0; o >>= 1) { if (tid < o) red[tid] += red[tid + o]; __syncthreads(); }
  const float mean = red[0] * (1.0f / CC);
  __syncthreads();
  float vs = 0.f;
  for (int c = tid; c < CC; c += 128) { float d = xr[c] - mean; vs += d * d; }
  red[tid] = vs; __syncthreads();
  for (int o = 64; o > 0; o >>= 1) { if (tid < o) red[tid] += red[tid + o]; __syncthreads(); }
  const float inv = rsqrtf(red[0] * (1.0f / CC) + 1e-5f);
  for (int c = tid; c < CC; c += 128)
    out[(size_t)row * CC + c] = (__bf16)((xr[c] - mean) * inv * w[c] + b[c]);
}

// kf[row,c] = max_k leaky_relu( Pg[idx[b,k,n], c] + Er[row, c] + bias[c] )
__global__ void knn_max_k(const float* __restrict__ Pg, const float* __restrict__ Er,
                          const int* __restrict__ idx, const float* __restrict__ bias,
                          __bf16* __restrict__ out)
{
  size_t t = (size_t)blockIdx.x * blockDim.x + threadIdx.x;
  if (t >= (size_t)MROWS * CC) return;
  const int c   = (int)(t % CC);
  const int row = (int)(t / CC);
  const int bb = row / NN, n = row % NN;
  const float base = Er[t] + bias[c];
  float m = -1e30f;
#pragma unroll
  for (int k = 0; k < KNN; ++k) {
    int g = idx[((size_t)bb * KNN + k) * NN + n];
    float v = Pg[(size_t)g * CC + c] + base;
    v = v > 0.f ? v : 0.2f * v;
    m = fmaxf(m, v);
  }
  out[t] = (__bf16)m;
}

__global__ void add_f32_k(const float* __restrict__ a, const float* __restrict__ b,
                          float* __restrict__ o, size_t n) {
  size_t t = (size_t)blockIdx.x * blockDim.x + threadIdx.x;
  if (t < n) o[t] = a[t] + b[t];
}
__global__ void gelu_bf16_k(const float* __restrict__ x, __bf16* __restrict__ o, size_t n) {
  size_t t = (size_t)blockIdx.x * blockDim.x + threadIdx.x;
  if (t < n) { float v = x[t]; o[t] = (__bf16)(0.5f * v * (1.0f + erff(v * 0.70710678f))); }
}
__global__ void cvt_bf16_k(const float* __restrict__ s, __bf16* __restrict__ d, size_t n) {
  size_t t = (size_t)blockIdx.x * blockDim.x + threadIdx.x;
  if (t < n) d[t] = (__bf16)s[t];
}
// Wd[c,j] = W[c, C+j] - W[c, j]  (W is [C, 2C] f32) -> bf16 [C, C]
__global__ void wdiff_bf16_k(const float* __restrict__ w, __bf16* __restrict__ d) {
  size_t t = (size_t)blockIdx.x * blockDim.x + threadIdx.x;
  if (t >= (size_t)CC * CC) return;
  int c = (int)(t / CC), j = (int)(t % CC);
  d[t] = (__bf16)(w[(size_t)c * (2 * CC) + CC + j] - w[(size_t)c * (2 * CC) + j]);
}
// dst[(bh*Nn + n)*64 + d] = src[(b*Nn + n)*stride + colOff + h*64 + d]
__global__ void split_heads_k(const float* __restrict__ src, int srcStride, int colOff,
                              __bf16* __restrict__ dst, int Nn) {
  size_t t = (size_t)blockIdx.x * blockDim.x + threadIdx.x;
  if (t >= (size_t)BHH * Nn * HD) return;
  int d = (int)(t % HD);
  size_t r = t / HD;
  int n = (int)(r % Nn);
  int bh = (int)(r / Nn);
  int b = bh / HH, h = bh % HH;
  dst[t] = (__bf16)src[((size_t)b * Nn + n) * srcStride + colOff + h * HD + d];
}
// transposed V: dst[(bh*64 + d)*Nn + n]
__global__ void split_headsT_k(const float* __restrict__ src, int srcStride, int colOff,
                               __bf16* __restrict__ dst, int Nn) {
  size_t t = (size_t)blockIdx.x * blockDim.x + threadIdx.x;
  if (t >= (size_t)BHH * Nn * HD) return;
  int n = (int)(t % Nn);
  size_t r = t / Nn;
  int d = (int)(r % HD);
  int bh = (int)(r / HD);
  int b = bh / HH, h = bh % HH;
  dst[t] = (__bf16)src[((size_t)b * Nn + n) * srcStride + colOff + h * HD + d];
}

// ---------------- host orchestration ----------------
extern "C" void kernel_launch(void* const* d_in, const int* in_sizes, int n_in,
                              void* d_out, int out_size, void* d_ws, size_t ws_size,
                              hipStream_t stream) {
  (void)in_sizes; (void)n_in; (void)out_size; (void)ws_size;
  const float* q_in      = (const float*)d_in[0];
  const float* v_in      = (const float*)d_in[1];
  const int*   idx_self  = (const int*)d_in[2];
  const int*   idx_cross = (const int*)d_in[3];
  const float* ln1_w = (const float*)d_in[4],  *ln1_b = (const float*)d_in[5];
  const float* qkv_w = (const float*)d_in[6];
  const float* aproj_w = (const float*)d_in[7], *aproj_b = (const float*)d_in[8];
  const float* lnq_w = (const float*)d_in[9],  *lnq_b = (const float*)d_in[10];
  const float* lnv_w = (const float*)d_in[11], *lnv_b = (const float*)d_in[12];
  const float* qm_w = (const float*)d_in[13], *km_w = (const float*)d_in[14], *vm_w = (const float*)d_in[15];
  const float* cproj_w = (const float*)d_in[16], *cproj_b = (const float*)d_in[17];
  const float* knn_w = (const float*)d_in[18], *knn_b = (const float*)d_in[19];
  const float* merge_w = (const float*)d_in[20], *merge_b = (const float*)d_in[21];
  const float* knnc_w = (const float*)d_in[22], *knnc_b = (const float*)d_in[23];
  const float* mergec_w = (const float*)d_in[24], *mergec_b = (const float*)d_in[25];
  const float* ln2_w = (const float*)d_in[26], *ln2_b = (const float*)d_in[27];
  const float* fc1_w = (const float*)d_in[28], *fc1_b = (const float*)d_in[29];
  const float* fc2_w = (const float*)d_in[30], *fc2_b = (const float*)d_in[31];

  const size_t E34 = (size_t)MROWS * CC;    // 3,145,728
  const size_t E15 = (size_t)MROWS * HIDD;  // 12,582,912

  size_t cur = 0;
  auto carveF = [&](size_t elems) -> float* {
    cur = (cur + 255) & ~(size_t)255; float* p = (float*)((char*)d_ws + cur); cur += elems * 4; return p; };
  auto carveB = [&](size_t elems) -> __bf16* {
    cur = (cur + 255) & ~(size_t)255; __bf16* p = (__bf16*)((char*)d_ws + cur); cur += elems * 2; return p; };

  float* f_big = carveF(E15);   // qkv / qh|kh|vh / fc1-out
  float* f_P   = carveF(E34);
  float* f_E   = carveF(E34);
  float* f_tmp = carveF(E34);
  float* f_q   = carveF(E34);   // running residual
  __bf16* b_x1 = carveB(E34);
  __bf16* b_x2 = carveB(E34);
  __bf16* b_x3 = carveB(E34);
  __bf16* b_x4 = carveB(E34);
  __bf16* b_big = carveB(E15);  // Qh | Kh | Vt, later gelu(h1)
  __bf16* Qh = b_big, *Khd = b_big + E34, *Vtd = b_big + 2 * E34;
  __bf16* wb_qkv   = carveB((size_t)3 * CC * CC);
  __bf16* wb_aproj = carveB((size_t)CC * CC);
  __bf16* wb_knn   = carveB((size_t)CC * 2 * CC);
  __bf16* wb_knnd  = carveB((size_t)CC * CC);
  __bf16* wb_merge = carveB((size_t)CC * 2 * CC);
  __bf16* wb_qm    = carveB((size_t)CC * CC);
  __bf16* wb_km    = carveB((size_t)CC * CC);
  __bf16* wb_vm    = carveB((size_t)CC * CC);
  __bf16* wb_cproj = carveB((size_t)CC * CC);
  __bf16* wb_knnc  = carveB((size_t)CC * 2 * CC);
  __bf16* wb_knncd = carveB((size_t)CC * CC);
  __bf16* wb_mergec= carveB((size_t)CC * 2 * CC);
  __bf16* wb_fc1   = carveB((size_t)HIDD * CC);
  __bf16* wb_fc2   = carveB((size_t)CC * HIDD);

  auto cvt = [&](const float* s, __bf16* d, size_t n) {
    cvt_bf16_k<<<dim3((unsigned)((n + 255) / 256)), dim3(256), 0, stream>>>(s, d, n);
  };
  auto gemm = [&](const __bf16* A, int lda, const __bf16* W, int ldw, const float* bias,
                  float* oF, __bf16* oB, int ldo, int Nc, int Kd, int acc) {
    gemm_bf16_k<<<dim3(MROWS / 128, Nc / 64), dim3(256), 0, stream>>>(
        A, lda, W, ldw, bias, oF, oB, ldo, MROWS, Nc, Kd, acc);
  };
  const unsigned g34 = (unsigned)((E34 + 255) / 256);

  // weight conversions
  cvt(qkv_w, wb_qkv, (size_t)3 * CC * CC);
  cvt(aproj_w, wb_aproj, (size_t)CC * CC);
  cvt(knn_w, wb_knn, (size_t)CC * 2 * CC);
  wdiff_bf16_k<<<dim3((unsigned)(((size_t)CC * CC + 255) / 256)), dim3(256), 0, stream>>>(knn_w, wb_knnd);
  cvt(merge_w, wb_merge, (size_t)CC * 2 * CC);
  cvt(qm_w, wb_qm, (size_t)CC * CC); cvt(km_w, wb_km, (size_t)CC * CC);
  cvt(vm_w, wb_vm, (size_t)CC * CC); cvt(cproj_w, wb_cproj, (size_t)CC * CC);
  cvt(knnc_w, wb_knnc, (size_t)CC * 2 * CC);
  wdiff_bf16_k<<<dim3((unsigned)(((size_t)CC * CC + 255) / 256)), dim3(256), 0, stream>>>(knnc_w, wb_knncd);
  cvt(mergec_w, wb_mergec, (size_t)CC * 2 * CC);
  cvt(fc1_w, wb_fc1, (size_t)HIDD * CC);
  cvt(fc2_w, wb_fc2, (size_t)CC * HIDD);

  // ---- stage 1: self attention + self kNN ----
  layernorm_k<<<dim3(MROWS), dim3(128), 0, stream>>>(q_in, ln1_w, ln1_b, b_x1);       // nq
  gemm(b_x1, CC, wb_qkv, CC, nullptr, f_big, nullptr, 3 * CC, 3 * CC, CC, 0);          // qkv
  split_heads_k <<<dim3(g34), dim3(256), 0, stream>>>(f_big, 3 * CC, 0,      Qh,  NN);
  split_heads_k <<<dim3(g34), dim3(256), 0, stream>>>(f_big, 3 * CC, CC,     Khd, NN);
  split_headsT_k<<<dim3(g34), dim3(256), 0, stream>>>(f_big, 3 * CC, 2 * CC, Vtd, NN);
  flash_attn_k<<<dim3(NN / 128, BHH), dim3(256), 0, stream>>>(Qh, Khd, Vtd, b_x2, NN, NN, 0.125f);
  gemm(b_x2, CC, wb_aproj, CC, aproj_b, nullptr, b_x3, CC, CC, CC, 0);                 // q1 (bf16)
  gemm(b_x1, CC, wb_knn, 2 * CC, nullptr, f_P, nullptr, CC, CC, CC, 0);                // P = nq@W1^T
  gemm(b_x1, CC, wb_knnd, CC, nullptr, f_E, nullptr, CC, CC, CC, 0);                   // E' = nq@(W2-W1)^T
  knn_max_k<<<dim3(g34), dim3(256), 0, stream>>>(f_P, f_E, idx_self, knn_b, b_x4);     // kf (bf16)
  gemm(b_x3, CC, wb_merge, 2 * CC, merge_b, f_tmp, nullptr, CC, CC, CC, 0);            // q1@M1^T + b
  gemm(b_x4, CC, wb_merge + CC, 2 * CC, nullptr, f_tmp, nullptr, CC, CC, CC, 1);       // += kf@M2^T
  add_f32_k<<<dim3(g34), dim3(256), 0, stream>>>(q_in, f_tmp, f_q, E34);               // q += q1

  // ---- stage 2: cross attention + cross kNN ----
  layernorm_k<<<dim3(MROWS), dim3(128), 0, stream>>>(f_q, lnq_w, lnq_b, b_x1);         // nq2
  layernorm_k<<<dim3(MROWS), dim3(128), 0, stream>>>(v_in, lnv_w, lnv_b, b_x2);        // nv
  gemm(b_x1, CC, wb_qm, CC, nullptr, f_big,            nullptr, CC, CC, CC, 0);        // qh
  gemm(b_x2, CC, wb_km, CC, nullptr, f_big + E34,      nullptr, CC, CC, CC, 0);        // kh
  gemm(b_x2, CC, wb_vm, CC, nullptr, f_big + 2 * E34,  nullptr, CC, CC, CC, 0);        // vh
  gemm(b_x2, CC, wb_knnc, 2 * CC, nullptr, f_P, nullptr, CC, CC, CC, 0);               // Pv = nv@W1c^T
  gemm(b_x1, CC, wb_knncd, CC, nullptr, f_E, nullptr, CC, CC, CC, 0);                  // E'c = nq2@(W2c-W1c)^T
  split_heads_k <<<dim3(g34), dim3(256), 0, stream>>>(f_big,           CC, 0, Qh,  NN);
  split_heads_k <<<dim3(g34), dim3(256), 0, stream>>>(f_big + E34,     CC, 0, Khd, NNK);
  split_headsT_k<<<dim3(g34), dim3(256), 0, stream>>>(f_big + 2 * E34, CC, 0, Vtd, NNK);
  flash_attn_k<<<dim3(NN / 128, BHH), dim3(256), 0, stream>>>(Qh, Khd, Vtd, b_x3, NN, NNK, 0.125f);
  gemm(b_x3, CC, wb_cproj, CC, cproj_b, nullptr, b_x2, CC, CC, CC, 0);                 // q2 (bf16)
  knn_max_k<<<dim3(g34), dim3(256), 0, stream>>>(f_P, f_E, idx_cross, knnc_b, b_x4);   // kf2
  gemm(b_x2, CC, wb_mergec, 2 * CC, mergec_b, f_tmp, nullptr, CC, CC, CC, 0);
  gemm(b_x4, CC, wb_mergec + CC, 2 * CC, nullptr, f_tmp, nullptr, CC, CC, CC, 1);
  add_f32_k<<<dim3(g34), dim3(256), 0, stream>>>(f_q, f_tmp, f_q, E34);                // q += q2

  // ---- stage 3: MLP ----
  layernorm_k<<<dim3(MROWS), dim3(128), 0, stream>>>(f_q, ln2_w, ln2_b, b_x1);         // h
  gemm(b_x1, CC, wb_fc1, CC, fc1_b, f_big, nullptr, HIDD, HIDD, CC, 0);                // fc1
  gelu_bf16_k<<<dim3((unsigned)((E15 + 255) / 256)), dim3(256), 0, stream>>>(f_big, b_big, E15);
  gemm(b_big, HIDD, wb_fc2, HIDD, fc2_b, f_tmp, nullptr, CC, CC, HIDD, 0);             // fc2
  add_f32_k<<<dim3(g34), dim3(256), 0, stream>>>(f_q, f_tmp, (float*)d_out, E34);      // out
}
